// GQAWithLoRA_62758062129376
// MI455X (gfx1250) — compile-verified, matrix operand order checked by reading, not run on hardware
//
#include <hip/hip_runtime.h>

#define DEV __device__ __forceinline__

constexpr int B_ = 4, S_ = 1024, D_ = 2048;
constexpr int H_ = 32, KVH_ = 8, HD_ = 64;
constexpr int KVD_ = HD_ * KVH_;      // 512
constexpr int RANK_ = 16;
constexpr float LORA_SCALE_ = 2.0f;   // ALPHA / RANK
constexpr int M_ = B_ * S_;           // 4096 rows

typedef __attribute__((ext_vector_type(16))) __bf16 v16bf;
typedef __attribute__((ext_vector_type(8)))  float  v8f;
typedef __attribute__((ext_vector_type(4)))  unsigned int v4u;
typedef __attribute__((ext_vector_type(8)))  int v8i;
typedef __attribute__((ext_vector_type(4)))  int v4i;

union FragU { v16bf v; unsigned int u[8]; };

DEV v8f vzero() {
  v8f c;
#pragma unroll
  for (int i = 0; i < 8; ++i) c[i] = 0.0f;
  return c;
}

// Load a 16x32 bf16 fragment (A-pattern) from row-major storage, leading dim ld.
// ISA layout: lanes 0-15 row=lane, K={0..7,16..23}; lanes 16-31 row=lane-16, K+=8.
DEV v16bf load_frag(const __bf16* p, int ld) {
  const int lane = threadIdx.x & 31;
  const int r = lane & 15;
  const int koff = (lane >> 4) << 3;   // 0 or 8
  const __bf16* q = p + r * ld + koff;
  FragU f;
#pragma unroll
  for (int j = 0; j < 8; ++j) {
    const int k = (j < 4) ? (2 * j) : (2 * j + 8);
    f.u[j] = *(const unsigned int*)(q + k);   // 2 packed bf16
  }
  return f.v;
}

DEV v8f wmma_bf16(v16bf a, v16bf b, v8f c) {
  return __builtin_amdgcn_wmma_f32_16x16x32_bf16(
      /*neg_a=*/false, a, /*neg_b=*/false, b,
      /*c_mod=*/(short)0, c, /*reuse_a=*/false, /*reuse_b=*/false);
}

// Async copy 16B from global to LDS (ASYNCcnt path, gfx1250).
DEV void async_copy_b128(unsigned lds_off, const __bf16* gptr) {
  asm volatile("global_load_async_to_lds_b128 %0, %1, off"
               :: "v"(lds_off), "v"(gptr)
               : "memory");
}

DEV void wait_async0() {
  asm volatile("s_wait_asynccnt 0x0" ::: "memory");
}

// TDM: DMA a 2D tile (tile_w x tile_h elements of 2 bytes) from a row-major
// tensor (row stride = stride_elems) into contiguous LDS at lds_off.
// D# per CDNA5 ISA ch.8: group0 = {count/lds/global/type}, group1 = dims.
// This toolchain exposes the 6-arg builtin: (v4u, v8i, v4i, v4i, v8i, i32 cpol).
DEV void tdm_load_2d_b16(unsigned lds_off, const void* gaddr,
                         unsigned tile_w, unsigned tile_h,
                         unsigned stride_elems,
                         unsigned tensor_w, unsigned tensor_h) {
  const unsigned long long ga = (unsigned long long)gaddr;
  v4u g0;
  g0[0] = 1u;                                   // count=1 (valid), user mode
  g0[1] = lds_off;                              // lds_addr (bytes)
  g0[2] = (unsigned)ga;                         // global_addr[31:0]
  g0[3] = (unsigned)(ga >> 32) | (2u << 30);    // global_addr[56:32] | type=2
  v8i g1;
  g1[0] = (int)(1u << 16);                      // wg_mask=0, data_size=1 (2B)
  g1[1] = (int)((tensor_w & 0xFFFFu) << 16);    // tensor_dim0[15:0] @63:48
  g1[2] = (int)((tensor_w >> 16) | ((tensor_h & 0xFFFFu) << 16)); // dim0 hi | dim1 lo
  g1[3] = (int)((tensor_h >> 16) | (tile_w << 16));               // dim1 hi | tile_dim0
  g1[4] = (int)(tile_h & 0xFFFFu);              // tile_dim1 (tile_dim2 = 0)
  g1[5] = (int)stride_elems;                    // tensor_dim0_stride[31:0]
  g1[6] = 0;                                    // stride hi | dim1_stride lo
  g1[7] = 0;
  v4i g2 = {0, 0, 0, 0};
  v4i g3 = {0, 0, 0, 0};
  v8i g4 = {0, 0, 0, 0, 0, 0, 0, 0};
  __builtin_amdgcn_tensor_load_to_lds(g0, g1, g2, g3, g4, /*cpol=*/0);
}

// ---------------- elementwise / prep kernels ----------------

__global__ void f32_to_bf16(const float* __restrict__ x, __bf16* __restrict__ y, int n) {
  int i = blockIdx.x * blockDim.x + threadIdx.x;
  if (i < n) y[i] = (__bf16)x[i];
}

// Wt[n][k] = W[k][n] + LORA_SCALE * sum_r lA[k][r]*lB[r][n], bf16, K = D_.
__global__ void prep_weight_lora_t(const float* __restrict__ W,
                                   const float* __restrict__ lA,
                                   const float* __restrict__ lB,
                                   __bf16* __restrict__ Wt, int N) {
  int idx = blockIdx.x * blockDim.x + threadIdx.x;
  if (idx >= N * D_) return;
  const int k = idx % D_;
  const int n = idx / D_;
  float acc = W[(size_t)k * N + n];
#pragma unroll
  for (int r = 0; r < RANK_; ++r)
    acc += LORA_SCALE_ * lA[k * RANK_ + r] * lB[r * N + n];
  Wt[(size_t)n * D_ + k] = (__bf16)acc;
}

// Wt[n][k] = W[k][n] (bf16), K = D_.
__global__ void prep_weight_t(const float* __restrict__ W, __bf16* __restrict__ Wt, int N) {
  int idx = blockIdx.x * blockDim.x + threadIdx.x;
  if (idx >= N * D_) return;
  const int k = idx % D_;
  const int n = idx / D_;
  Wt[(size_t)n * D_ + k] = (__bf16)W[(size_t)k * N + n];
}

// In-place RoPE on bf16 tensor laid out (B, S, nh, HD); rowd = nh*HD.
__global__ void rope_kernel(__bf16* __restrict__ T,
                            const float* __restrict__ fcos,
                            const float* __restrict__ fsin,
                            int nh, int rowd) {
  int idx = blockIdx.x * blockDim.x + threadIdx.x;
  const int total = B_ * S_ * nh * (HD_ / 2);
  if (idx >= total) return;
  const int i2 = idx & 31;            // HD_/2 == 32
  int tmp = idx >> 5;
  const int h = tmp % nh;  tmp /= nh;
  const int s = tmp % S_;
  const int b = tmp / S_;
  const float c = fcos[s * 32 + i2];
  const float sn = fsin[s * 32 + i2];
  __bf16* p = T + ((size_t)(b * S_ + s) * rowd + h * HD_ + 2 * i2);
  const float x0 = (float)p[0], x1 = (float)p[1];
  p[0] = (__bf16)(x0 * c - x1 * sn);
  p[1] = (__bf16)(x0 * sn + x1 * c);
}

// Vt[b][g][hd][t] = V[b][t][g][hd]
__global__ void transpose_v(const __bf16* __restrict__ V, __bf16* __restrict__ Vt) {
  int idx = blockIdx.x * blockDim.x + threadIdx.x;
  if (idx >= B_ * S_ * KVD_) return;
  const int hd = idx % HD_;
  const int g = (idx / HD_) % KVH_;
  const int t = (idx / KVD_) % S_;
  const int b = idx / (S_ * KVD_);
  Vt[((size_t)((b * KVH_ + g) * HD_ + hd)) * S_ + t] = V[idx];
}

// ---------------- WMMA GEMM: C[M][N] = A[M][K] * Bt[N][K]^T ----------------
// Double-buffered LDS staging: A via GLOBAL_LOAD_ASYNC_TO_LDS_B128 (ASYNCcnt),
// B (weights) via a single TDM tensor_load_to_lds per step (TENSORcnt).

template <typename OutT>
__global__ __launch_bounds__(256)
void gemm_bt_wmma(const __bf16* __restrict__ A, const __bf16* __restrict__ Bt,
                  OutT* __restrict__ C, int M, int N, int K) {
  __shared__ __align__(16) __bf16 sA[2][128 * 32];
  __shared__ __align__(16) __bf16 sB[2][128 * 32];
  const int bm = blockIdx.y * 128;
  const int bn = blockIdx.x * 128;
  const int t = threadIdx.x;
  const int lane = t & 31;
  const int w = t >> 5;
  const int wr = (w >> 1) * 32;   // wave row origin in tile
  const int wc = (w & 1) * 64;    // wave col origin in tile

  // A tile: per-lane async copies (8 KB / step across 256 lanes).
  auto issueA = [&](int buf, int k0) {
#pragma unroll
    for (int i = 0; i < 2; ++i) {
      const int idx = t + i * 256;       // 0..511
      const int row = idx >> 2;          // 0..127
      const int ch = (idx & 3) * 8;      // 16-byte chunk
      const unsigned aoff = (unsigned)(size_t)&sA[buf][row * 32 + ch];
      async_copy_b128(aoff, &A[(size_t)(bm + row) * K + k0 + ch]);
    }
  };
  // B tile: one TDM 2D descriptor (128 rows x 32 cols bf16, row stride K).
  auto issueB = [&](int buf, int k0) {
    tdm_load_2d_b16((unsigned)(size_t)&sB[buf][0],
                    &Bt[(size_t)bn * K + k0],
                    /*tile_w=*/32, /*tile_h=*/128,
                    /*stride=*/(unsigned)K,
                    /*tensor_w=*/(unsigned)K, /*tensor_h=*/(unsigned)N);
  };

  v8f acc[2][4];
#pragma unroll
  for (int mi = 0; mi < 2; ++mi)
#pragma unroll
    for (int ni = 0; ni < 4; ++ni) acc[mi][ni] = vzero();

  const int nk = K / 32;
  issueA(0, 0);
  if (w == 0) issueB(0, 0);
  int cur = 0;
  for (int i = 0; i < nk; ++i) {
    wait_async0();                              // my A fills done
    __builtin_amdgcn_s_wait_tensorcnt(0);       // wave0's B fill done (no-op elsewhere)
    __syncthreads();                            // publish tile, retire prior reads
    if (i + 1 < nk) {
      issueA(cur ^ 1, (i + 1) * 32);
      if (w == 0) issueB(cur ^ 1, (i + 1) * 32);
    }

    v16bf af[2], bfr[4];
#pragma unroll
    for (int mi = 0; mi < 2; ++mi) af[mi] = load_frag(&sA[cur][(wr + mi * 16) * 32], 32);
#pragma unroll
    for (int ni = 0; ni < 4; ++ni) bfr[ni] = load_frag(&sB[cur][(wc + ni * 16) * 32], 32);
#pragma unroll
    for (int mi = 0; mi < 2; ++mi)
#pragma unroll
      for (int ni = 0; ni < 4; ++ni)
        acc[mi][ni] = wmma_bf16(af[mi], bfr[ni], acc[mi][ni]);
    cur ^= 1;
  }

  const int gg = lane >> 4, nn = lane & 15;
#pragma unroll
  for (int mi = 0; mi < 2; ++mi)
#pragma unroll
    for (int ni = 0; ni < 4; ++ni)
#pragma unroll
      for (int r = 0; r < 8; ++r) {
        const int m = bm + wr + mi * 16 + r + 8 * gg;
        const int n = bn + wc + ni * 16 + nn;
        C[(size_t)m * N + n] = (OutT)acc[mi][ni][r];
      }
}

// ---------------- Flash attention (causal, GQA rep=4), bf16 WMMA ----------------
// Q: (B,S,H,HD) bf16 (post-RoPE); K: (B,S,KVH,HD) bf16 (post-RoPE);
// Vt: (B,KVH,HD,S) bf16; ctx out: (B,S,H,HD) bf16 (== (M, D) row-major).

__global__ __launch_bounds__(256)
void flash_attn(const __bf16* __restrict__ Q, const __bf16* __restrict__ Kt,
                const __bf16* __restrict__ Vt, __bf16* __restrict__ ctx) {
  __shared__ __align__(16) __bf16 sP[8][16 * 32];   // per-wave P tile
  const int lane = threadIdx.x & 31;
  const int w = threadIdx.x >> 5;
  const int h = blockIdx.y;
  const int b = blockIdx.z;
  const int q0 = blockIdx.x * 128 + w * 16;   // 16 query rows per wave
  const int g = h >> 2;                       // kv head (rep = 4)
  const int gg = lane >> 4;
  const int nn = lane & 15;

  const __bf16* qbase = Q + ((size_t)(b * S_ + q0) * D_ + h * HD_);
  const v16bf qf0 = load_frag(qbase, D_);        // hd 0..31
  const v16bf qf1 = load_frag(qbase + 32, D_);   // hd 32..63

  v8f o[4];
  float rmax[8], rsum[8];
#pragma unroll
  for (int ni = 0; ni < 4; ++ni) o[ni] = vzero();
#pragma unroll
  for (int r = 0; r < 8; ++r) { rmax[r] = -1e30f; rsum[r] = 0.0f; }

  const int nkb = ((q0 + 15) >> 5) + 1;   // causal: 32-wide key blocks
  for (int kb = 0; kb < nkb; ++kb) {
    const int t0 = kb * 32;
    v8f sc[2];
#pragma unroll
    for (int half = 0; half < 2; ++half) {
      const __bf16* kbase = Kt + ((size_t)(b * S_ + t0 + half * 16) * KVD_ + g * HD_);
      const v16bf kf0 = load_frag(kbase, KVD_);
      const v16bf kf1 = load_frag(kbase + 32, KVD_);
      v8f c = vzero();
      c = wmma_bf16(qf0, kf0, c);
      c = wmma_bf16(qf1, kf1, c);
      sc[half] = c;
    }
    __bf16* sp = &sP[w][0];
#pragma unroll
    for (int r = 0; r < 8; ++r) {
      const int row = q0 + r + 8 * gg;
      float s0 = sc[0][r] * 0.125f;          // 1/sqrt(64)
      float s1 = sc[1][r] * 0.125f;
      if (t0 + nn > row)      s0 = -1e30f;   // causal mask
      if (t0 + 16 + nn > row) s1 = -1e30f;
      float lm = fmaxf(s0, s1);
#pragma unroll
      for (int m = 1; m < 16; m <<= 1) lm = fmaxf(lm, __shfl_xor(lm, m, 32));
      const float nm = fmaxf(rmax[r], lm);
      const float corr = __expf(rmax[r] - nm);
      rmax[r] = nm;
      const float e0 = __expf(s0 - nm);
      const float e1 = __expf(s1 - nm);
      float ls = e0 + e1;
#pragma unroll
      for (int m = 1; m < 16; m <<= 1) ls += __shfl_xor(ls, m, 32);
      rsum[r] = rsum[r] * corr + ls;
#pragma unroll
      for (int ni = 0; ni < 4; ++ni) o[ni][r] *= corr;
      sp[(r + 8 * gg) * 32 + nn] = (__bf16)e0;
      sp[(r + 8 * gg) * 32 + 16 + nn] = (__bf16)e1;
    }
    __builtin_amdgcn_wave_barrier();          // C-layout -> A-layout via LDS
    const v16bf pf = load_frag(sp, 32);
    __builtin_amdgcn_wave_barrier();
#pragma unroll
    for (int ni = 0; ni < 4; ++ni) {
      const __bf16* vb = Vt + ((size_t)((b * KVH_ + g) * HD_ + ni * 16)) * S_ + t0;
      const v16bf vf = load_frag(vb, S_);
      o[ni] = wmma_bf16(pf, vf, o[ni]);
    }
  }

#pragma unroll
  for (int r = 0; r < 8; ++r) rsum[r] = 1.0f / rsum[r];
#pragma unroll
  for (int ni = 0; ni < 4; ++ni)
#pragma unroll
    for (int r = 0; r < 8; ++r) {
      const int s = q0 + r + 8 * gg;
      const int hd = ni * 16 + nn;
      ctx[(size_t)(b * S_ + s) * D_ + h * HD_ + hd] = (__bf16)(o[ni][r] * rsum[r]);
    }
}

// ---------------- host launcher ----------------

extern "C" void kernel_launch(void* const* d_in, const int* in_sizes, int n_in,
                              void* d_out, int out_size, void* d_ws, size_t ws_size,
                              hipStream_t stream) {
  const float* x  = (const float*)d_in[0];
  const float* fc = (const float*)d_in[1];
  const float* fs = (const float*)d_in[2];
  const float* Wq = (const float*)d_in[3];
  const float* Wk = (const float*)d_in[4];
  const float* Wv = (const float*)d_in[5];
  const float* Wo = (const float*)d_in[6];
  const float* qA = (const float*)d_in[7];
  const float* qB = (const float*)d_in[8];
  const float* kA = (const float*)d_in[9];
  const float* kB = (const float*)d_in[10];
  const float* vA = (const float*)d_in[11];
  const float* vB = (const float*)d_in[12];
  float* out = (float*)d_out;

  char* p = (char*)d_ws;
  auto alloc = [&](size_t bytes) {
    char* r = p;
    p += (bytes + 255) & ~(size_t)255;
    return r;
  };
  __bf16* Xb  = (__bf16*)alloc((size_t)M_ * D_ * 2);
  __bf16* Wqt = (__bf16*)alloc((size_t)D_ * D_ * 2);
  __bf16* Wkt = (__bf16*)alloc((size_t)KVD_ * D_ * 2);
  __bf16* Wvt = (__bf16*)alloc((size_t)KVD_ * D_ * 2);
  __bf16* Wot = (__bf16*)alloc((size_t)D_ * D_ * 2);
  __bf16* Qb  = (__bf16*)alloc((size_t)M_ * D_ * 2);
  __bf16* Kb  = (__bf16*)alloc((size_t)M_ * KVD_ * 2);
  __bf16* Vb  = (__bf16*)alloc((size_t)M_ * KVD_ * 2);
  __bf16* Vtb = (__bf16*)alloc((size_t)M_ * KVD_ * 2);
  __bf16* Ctx = (__bf16*)alloc((size_t)M_ * D_ * 2);

  auto nblk = [](long n) { return (unsigned)((n + 255) / 256); };

  f32_to_bf16<<<nblk((long)M_ * D_), 256, 0, stream>>>(x, Xb, M_ * D_);

  prep_weight_lora_t<<<nblk((long)D_ * D_),   256, 0, stream>>>(Wq, qA, qB, Wqt, D_);
  prep_weight_lora_t<<<nblk((long)KVD_ * D_), 256, 0, stream>>>(Wk, kA, kB, Wkt, KVD_);
  prep_weight_lora_t<<<nblk((long)KVD_ * D_), 256, 0, stream>>>(Wv, vA, vB, Wvt, KVD_);
  prep_weight_t<<<nblk((long)D_ * D_), 256, 0, stream>>>(Wo, Wot, D_);

  dim3 gq(D_ / 128, M_ / 128);
  gemm_bt_wmma<__bf16><<<gq, 256, 0, stream>>>(Xb, Wqt, Qb, M_, D_, D_);
  dim3 gkv(KVD_ / 128, M_ / 128);
  gemm_bt_wmma<__bf16><<<gkv, 256, 0, stream>>>(Xb, Wkt, Kb, M_, KVD_, D_);
  gemm_bt_wmma<__bf16><<<gkv, 256, 0, stream>>>(Xb, Wvt, Vb, M_, KVD_, D_);

  rope_kernel<<<nblk((long)B_ * S_ * H_ * 32),   256, 0, stream>>>(Qb, fc, fs, H_, D_);
  rope_kernel<<<nblk((long)B_ * S_ * KVH_ * 32), 256, 0, stream>>>(Kb, fc, fs, KVH_, KVD_);
  transpose_v<<<nblk((long)B_ * S_ * KVD_), 256, 0, stream>>>(Vb, Vtb);

  dim3 gf(S_ / 128, H_, B_);
  flash_attn<<<gf, 256, 0, stream>>>(Qb, Kb, Vtb, Ctx);

  dim3 go(D_ / 128, M_ / 128);
  gemm_bt_wmma<float><<<go, 256, 0, stream>>>(Ctx, Wot, out, M_, D_, D_);
}
